// FullFloodFillNetworkv2_609885356698
// MI455X (gfx1250) — matrix operand order repaired and checked by reading, not compiled, not run on hardware
//
#include <hip/hip_runtime.h>
#include <hip/hip_bf16.h>
#include <math.h>

// ---------------------------------------------------------------------------
// FullFloodFillNetworkv2 for MI455X (gfx1250, wave32, WMMA)
// Persistent f16 K/V cache; per-ring flash attention with
// v_wmma_f32_16x16x32_f16; all fragments loaded as contiguous 32B vectors.
// K cache: Kh[b][h][t][d]  (row t contiguous -> score B-frag = 1 v16h load)
// V cache: VhT[b][h][d][t] (row d contiguous -> PV B-frag = 1 v16h load)
// LDS activation tiles stored [col][128ch]    -> proj B-frag = 1 v16h load
// ---------------------------------------------------------------------------

typedef _Float16 h16;
typedef __attribute__((ext_vector_type(8)))  _Float16 v8h;
typedef __attribute__((ext_vector_type(16))) _Float16 v16h;
typedef __attribute__((ext_vector_type(8)))  float    v8f;

#define BB 4
#define CC 128
#define HH 8
#define GG 64
#define FF 4096     // G*G
#define DH 16       // C/H

// ---- WMMA fragment index maps (wave32; contiguous K order is this code's choice) -------------------
// A-frag 16x32 f16: lane -> M=lane&15, K = hs*16 + j  (fully contiguous)
//   => slots 0..15 are contiguous K starting at  hs*16
//      (same K order as the B-frag)
// B-frag 32x16 f16: lane -> N=lane&15, K = hs*16 + j  (fully contiguous)
// C/D-frag 16x16 f32: VGPR r -> M = r + 8*hs, N = lane&15
__device__ __forceinline__ int kidxA(int hs, int j) {
  return hs * 16 + j;
}
__device__ __forceinline__ int kidxB(int hs, int j) {
  return hs * 16 + j;
}

__device__ __forceinline__ v8f wmma_f16(v16h a, v16h b, v8f c) {
  return __builtin_amdgcn_wmma_f32_16x16x32_f16(false, a, false, b,
                                                (short)0, c, false, false);
}
__device__ __forceinline__ v16h cat8(v8h lo, v8h hi) {
  return __builtin_shufflevector(lo, hi, 0, 1, 2, 3, 4, 5, 6, 7, 8, 9, 10, 11,
                                 12, 13, 14, 15);
}

// Load A-frag (16 rows, one K-step of 32) from a row-major fp32 [*,CC] weight.
// One contiguous 16-float run per lane; compiler vectorizes to b128 loads.
__device__ __forceinline__ v16h load_wA(const float* __restrict__ W,
                                        int row0, int kb, int lane) {
  int m = lane & 15, hs = lane >> 4;
  const float* wr = W + (size_t)(row0 + m) * CC + kb;
  v16h a;
#pragma unroll
  for (int j = 0; j < 16; ++j) a[j] = (h16)wr[kidxA(hs, j)];
  return a;
}

// B-frag from transposed LDS activation tile act[col][CC]: one v16h load.
__device__ __forceinline__ v16h load_bB(const h16* __restrict__ act, int kb,
                                        int n0, int lane) {
  int n = n0 + (lane & 15), hs = lane >> 4;
  return *(const v16h*)(act + (size_t)n * CC + kb + hs * 16);
}

// ---------------------------------------------------------------------------
// feats = x ; scores = 0
__global__ __launch_bounds__(256) void copy_init_kernel(
    const float* __restrict__ x, float* __restrict__ feats,
    float* __restrict__ score) {
  size_t i = (size_t)blockIdx.x * blockDim.x + threadIdx.x;
  if (i < (size_t)BB * CC * FF) feats[i] = x[i];
  if (i < (size_t)BB * FF) score[i] = 0.f;
}

// ---------------------------------------------------------------------------
// Initial K/V cache from x. Grid (B,H), 128 threads = 4 waves.
__global__ __launch_bounds__(128) void kv_init_kernel(
    const float* __restrict__ x, const float* __restrict__ Wk0,
    const float* __restrict__ Wv0, h16* __restrict__ Kh,
    h16* __restrict__ VhT) {
  int b = blockIdx.x, h = blockIdx.y;
  int tid = threadIdx.x, lane = tid & 31, w = tid >> 5;
  int m = lane & 15, hs = lane >> 4;

  v16h aK[4], aV[4];
#pragma unroll
  for (int s = 0; s < 4; ++s) {
    aK[s] = load_wA(Wk0, h * DH, s * 32, lane);
    aV[s] = load_wA(Wv0, h * DH, s * 32, lane);
  }
  const float* xb = x + (size_t)b * CC * FF;
  size_t hb = (size_t)(b * HH + h) * FF;
  h16* vt = VhT + (size_t)(b * HH + h) * DH * FF;

  for (int t0 = w * 16; t0 < FF; t0 += 64) {
    int t = t0 + m;
    v8f cK = {}, cV = {};
#pragma unroll
    for (int s = 0; s < 4; ++s) {
      v16h bx;
#pragma unroll
      for (int j = 0; j < 16; ++j) {
        int c = s * 32 + kidxB(hs, j);
        bx[j] = (h16)xb[(size_t)c * FF + t];
      }
      cK = wmma_f16(aK[s], bx, cK);
      cV = wmma_f16(aV[s], bx, cV);
    }
    size_t base = (hb + t) * DH;
#pragma unroll
    for (int rr = 0; rr < 8; ++rr) {
      int d = rr + hs * 8;
      Kh[base + d] = (h16)cK[rr];
      vt[(size_t)d * FF + t] = (h16)cV[rr];
    }
  }
}

// ---------------------------------------------------------------------------
// Per-ring attention: q = Wq0 @ feats[:, ring]; flash attention over cached
// 4096 keys; result (with residual) staged into newbf.
__global__ __launch_bounds__(128) void ring_kernel(
    const float* __restrict__ feats, const float* __restrict__ Wq0,
    const h16* __restrict__ Kh, const h16* __restrict__ VhT,
    float* __restrict__ newbf, int r) {
  int b = blockIdx.x, h = blockIdx.y;
  int tid = threadIdx.x, lane = tid & 31, w = tid >> 5;
  int m = lane & 15, hs = lane >> 4;
  int lo = (r - (GG - 1) > 0) ? r - (GG - 1) : 0;
  int hi = (r < GG - 1) ? r : GG - 1;
  int nq = hi - lo + 1;

  __shared__ alignas(32) h16 bf[64 * CC];       // [col][ch], 16KB
  __shared__ alignas(32) h16 qlds[64 * DH];     // [i][d], pre-scaled
  __shared__ alignas(32) h16 plds[4][16 * 32];  // per-wave P tile [row][t]

  const float* fb = feats + (size_t)b * CC * FF;

  for (int idx = tid; idx < 64 * CC; idx += 128) {
    int i = idx >> 7, c = idx & 127;
    float v = 0.f;
    if (i < nq) {
      int col = (lo + i) * GG + (r - lo - i);
      v = fb[(size_t)c * FF + col];
    }
    bf[idx] = (h16)v;
  }
  __syncthreads();

  // ---- Q projection: M=d (Wq0 rows of head h), N=i, K=128 -----------------
  {
    v8f cq = {};
#pragma unroll
    for (int s = 0; s < 4; ++s) {
      v16h aw = load_wA(Wq0, h * DH, s * 32, lane);
      v16h bb = load_bB(bf, s * 32, w * 16, lane);
      cq = wmma_f16(aw, bb, cq);
    }
    int i = w * 16 + m;
#pragma unroll
    for (int rr = 0; rr < 8; ++rr) {
      int d = rr + hs * 8;
      qlds[i * DH + d] = (h16)(cq[rr] * 0.25f);   // fold 1/sqrt(DH)
    }
  }
  __syncthreads();

  // ---- Flash attention: wave w owns query rows i0..i0+15 ------------------
  int i0 = w * 16;
  v8h zero8 = {};
  v16h aq = cat8(*(const v8h*)(qlds + (i0 + m) * DH), *(const v8h*)(qlds + (i0 + m) * DH + 8));

  v8f acc = {};
  float rm[8], rl[8];
#pragma unroll
  for (int rr = 0; rr < 8; ++rr) { rm[rr] = -1e30f; rl[rr] = 0.f; }

  const h16* Kb = Kh + (size_t)(b * HH + h) * FF * DH;
  const h16* VbT = VhT + (size_t)(b * HH + h) * DH * FF;
  const v16h zer = {};

#pragma unroll 2
  for (int t0 = 0; t0 < FF; t0 += 32) {
    __builtin_prefetch(Kb + (size_t)(t0 + 32) * DH, 0, 3);

    // K cache [t][d] -> one contiguous v16h per lane (hs==0 lanes carry data)
    v16h k0v = *(const v16h*)(Kb + (size_t)(t0 + m) * DH);
    v16h k1v = *(const v16h*)(Kb + (size_t)(t0 + 16 + m) * DH);
    v16h bk0 = hs ? zer : k0v;
    v16h bk1 = hs ? zer : k1v;

    v8f cz0 = {}, cz1 = {};
    v8f s0 = wmma_f16(aq, bk0, cz0);
    v8f s1 = wmma_f16(aq, bk1, cz1);

#pragma unroll
    for (int rr = 0; rr < 8; ++rr) {
      float x0 = s0[rr], x1 = s1[rr];
      float mx = fmaxf(x0, x1);
#pragma unroll
      for (int off = 1; off < 16; off <<= 1)
        mx = fmaxf(mx, __shfl_xor(mx, off, 16));
      float mn = fmaxf(rm[rr], mx);
      float sc = __expf(rm[rr] - mn);
      float p0 = __expf(x0 - mn);
      float p1 = __expf(x1 - mn);
      float rs = p0 + p1;
#pragma unroll
      for (int off = 1; off < 16; off <<= 1)
        rs += __shfl_xor(rs, off, 16);
      rl[rr] = rl[rr] * sc + rs;
      rm[rr] = mn;
      acc[rr] *= sc;
      int il = rr + hs * 8;                    // local row
      plds[w][il * 32 + m] = (h16)p0;          // t_local = m
      plds[w][il * 32 + 16 + m] = (h16)p1;     // t_local = 16+m
    }
    // same-wave LDS exchange; per-wave DS ordering + compiler DScnt waits
    const h16* prow = plds[w] + m * 32;
    v16h ap = *(const v16h*)(prow + hs * 16);
    // V cache [d][t] -> one contiguous v16h per lane (N=d=m, K=j->t)
    v16h bv = *(const v16h*)(VbT + (size_t)m * FF + t0 + hs * 16);
    acc = wmma_f16(ap, bv, acc);               // O += P @ V
  }

  // ---- finalize: o/l + residual -> newbf ----------------------------------
#pragma unroll
  for (int rr = 0; rr < 8; ++rr) {
    int i = i0 + rr + hs * 8;                  // query index
    int d = m;                                 // head-dim index
    if (i < nq) {
      int col = (lo + i) * GG + (r - lo - i);
      float res = fb[(size_t)(h * DH + d) * FF + col];
      newbf[((size_t)b * CC + h * DH + d) * 64 + i] = acc[rr] / rl[rr] + res;
    }
  }
}

// ---------------------------------------------------------------------------
// Commit ring columns to feats and update K/V cache columns via WMMA GEMM.
__global__ __launch_bounds__(128) void kv_update_kernel(
    float* __restrict__ feats, const float* __restrict__ Wk0,
    const float* __restrict__ Wv0, const float* __restrict__ newbf,
    h16* __restrict__ Kh, h16* __restrict__ VhT, int r) {
  int b = blockIdx.x, h = blockIdx.y;
  int tid = threadIdx.x, lane = tid & 31, w = tid >> 5;
  int m = lane & 15, hs = lane >> 4;
  int lo = (r - (GG - 1) > 0) ? r - (GG - 1) : 0;
  int hi = (r < GG - 1) ? r : GG - 1;
  int nq = hi - lo + 1;

  __shared__ alignas(32) h16 nb[64 * CC];      // [col][ch]

  for (int idx = tid; idx < 64 * CC; idx += 128) {
    int i = idx >> 7, c = idx & 127;
    float v = 0.f;
    if (i < nq) {
      v = newbf[((size_t)b * CC + c) * 64 + i];
      if ((c >> 4) == h) {                     // this block owns channel slice
        int col = (lo + i) * GG + (r - lo - i);
        feats[(size_t)b * CC * FF + (size_t)c * FF + col] = v;
      }
    }
    nb[idx] = (h16)v;
  }
  __syncthreads();

  v8f cK = {}, cV = {};
#pragma unroll
  for (int s = 0; s < 4; ++s) {
    v16h aK = load_wA(Wk0, h * DH, s * 32, lane);
    v16h aV = load_wA(Wv0, h * DH, s * 32, lane);
    v16h bb = load_bB(nb, s * 32, w * 16, lane);
    cK = wmma_f16(aK, bb, cK);
    cV = wmma_f16(aV, bb, cV);
  }
  int i = w * 16 + m;
  if (i < nq) {
    int col = (lo + i) * GG + (r - lo - i);
    size_t kbase = ((size_t)(b * HH + h) * FF + col) * DH;
    h16* vt = VhT + (size_t)(b * HH + h) * DH * FF;
#pragma unroll
    for (int rr = 0; rr < 8; ++rr) {
      int d = rr + hs * 8;
      Kh[kbase + d] = (h16)cK[rr];
      vt[(size_t)d * FF + col] = (h16)cV[rr];
    }
  }
}

// ---------------------------------------------------------------------------
// Anchor attend collapses: softmax over one key == 1 => o = Wv0 @ feats[:,0].
// Also normalized k1 = normalize_per_head(Wk1 @ feats[:,0]).
__global__ __launch_bounds__(128) void anchor_kernel(
    const float* __restrict__ feats, const float* __restrict__ Wv0,
    const float* __restrict__ Wk1, float* __restrict__ va,
    float* __restrict__ k1n) {
  int tid = threadIdx.x;
  for (int idx = tid; idx < BB * CC; idx += blockDim.x) {
    int b = idx / CC, c = idx % CC;
    const float* fb = feats + (size_t)b * CC * FF;   // anchor column = 0
    float sv = 0.f, sk = 0.f;
    for (int cc = 0; cc < CC; ++cc) {
      float f = fb[(size_t)cc * FF];
      sv += Wv0[(size_t)c * CC + cc] * f;
      sk += Wk1[(size_t)c * CC + cc] * f;
    }
    va[idx] = sv;
    k1n[idx] = sk;
  }
  __syncthreads();
  for (int idx = tid; idx < BB * HH; idx += blockDim.x) {
    int base = idx * DH;
    float ssq = 0.f;
    for (int d = 0; d < DH; ++d) { float v = k1n[base + d]; ssq += v * v; }
    float s = rsqrtf(ssq + 1e-8f);
    for (int d = 0; d < DH; ++d) k1n[base + d] *= s;
  }
}

// ---------------------------------------------------------------------------
// Cosine scoring: q1 = Wq1 @ (x + va); m = cos(q1_h, k1_h); mean over heads.
// Grid (B, F/64); each block loops over 8 heads re-using the LDS qx tile.
__global__ __launch_bounds__(128) void score_kernel(
    const float* __restrict__ x, const float* __restrict__ Wq1,
    const float* __restrict__ va, const float* __restrict__ k1n,
    float* __restrict__ outscore) {
  int b = blockIdx.x;
  int n0 = blockIdx.y * 64;
  int tid = threadIdx.x, lane = tid & 31, w = tid >> 5;
  int m = lane & 15, hs = lane >> 4;

  __shared__ alignas(32) h16 qx[64 * CC];      // [col][ch]
  __shared__ float sacc[64];

  for (int idx = tid; idx < 64 * CC; idx += 128) {
    int i = idx >> 7, c = idx & 127;
    qx[idx] = (h16)(x[(size_t)b * CC * FF + (size_t)c * FF + n0 + i] +
                    va[b * CC + c]);
  }
  if (tid < 64) sacc[tid] = 0.f;
  __syncthreads();

  for (int h = 0; h < HH; ++h) {
    v8f cq = {};
#pragma unroll
    for (int s = 0; s < 4; ++s) {
      v16h aw = load_wA(Wq1, h * DH, s * 32, lane);
      v16h bb = load_bB(qx, s * 32, w * 16, lane);
      cq = wmma_f16(aw, bb, cq);
    }
    float ssq = 0.f, dot = 0.f;
#pragma unroll
    for (int rr = 0; rr < 8; ++rr) {
      float qv = cq[rr];
      ssq += qv * qv;
      dot += qv * k1n[(b * HH + h) * DH + rr + hs * 8];
    }
    ssq += __shfl_xor(ssq, 16, 32);   // combine complementary d-halves
    dot += __shfl_xor(dot, 16, 32);
    float mm = dot * rsqrtf(ssq + 1e-8f);
    if (hs == 0) sacc[w * 16 + m] += (mm + 1.f) * 0.5f * (1.f / HH);
  }
  __syncthreads();
  if (tid < 64) outscore[(size_t)b * FF + n0 + tid] = sacc[tid];
}

// ---------------------------------------------------------------------------
extern "C" void kernel_launch(void* const* d_in, const int* in_sizes, int n_in,
                              void* d_out, int out_size, void* d_ws,
                              size_t ws_size, hipStream_t stream) {
  (void)in_sizes; (void)n_in; (void)out_size; (void)ws_size;
  const float* x   = (const float*)d_in[0];
  const float* Wq0 = (const float*)d_in[1];
  const float* Wk0 = (const float*)d_in[2];
  const float* Wv0 = (const float*)d_in[3];
  const float* Wq1 = (const float*)d_in[4];
  const float* Wk1 = (const float*)d_in[5];
  // d_in[6] rings, d_in[7] ring_sizes: ring geometry reproduced analytically.

  float* feats = (float*)d_out;                        // B*C*F
  float* score = feats + (size_t)BB * CC * FF;         // B*F

  char* ws = (char*)d_ws;
  h16* Kh  = (h16*)ws;                                 // [b][h][t][d]  4 MB
  h16* VhT = Kh + (size_t)BB * HH * FF * DH;           // [b][h][d][t]  4 MB
  float* newbf = (float*)(VhT + (size_t)BB * HH * FF * DH);  // B*C*64 f32
  float* va  = newbf + (size_t)BB * CC * 64;           // B*C
  float* k1n = va + BB * CC;                           // B*C

  copy_init_kernel<<<(BB * CC * FF + 255) / 256, 256, 0, stream>>>(x, feats,
                                                                   score);
  kv_init_kernel<<<dim3(BB, HH), 128, 0, stream>>>(x, Wk0, Wv0, Kh, VhT);

  for (int r = 0; r < 2 * GG - 1; ++r) {
    ring_kernel<<<dim3(BB, HH), 128, 0, stream>>>(feats, Wq0, Kh, VhT, newbf,
                                                  r);
    kv_update_kernel<<<dim3(BB, HH), 128, 0, stream>>>(feats, Wk0, Wv0, newbf,
                                                       Kh, VhT, r);
  }

  anchor_kernel<<<1, 128, 0, stream>>>(feats, Wv0, Wk1, va, k1n);
  score_kernel<<<dim3(BB, FF / 64), 128, 0, stream>>>(x, Wq1, va, k1n, score);
}
